// GQAAttn_3023656976648
// MI455X (gfx1250) — compile-verified, matrix operand order checked by reading, not run on hardware
//
#include <hip/hip_runtime.h>
#include <hip/hip_bf16.h>
#include <stdint.h>

// ---------------------------------------------------------------------------
// Problem constants (from the reference)
// ---------------------------------------------------------------------------
#define BB   2
#define TT   2048
#define DD   1024
#define HH   16
#define HKV  4
#define DH   64
#define NQKV 1536           // H*DH + 2*HKV*DH = 1024 + 256 + 256
#define MROWS (BB*TT)       // 4096

typedef __attribute__((ext_vector_type(16))) __bf16 v16bf;
typedef __attribute__((ext_vector_type(8)))  float  v8f;
typedef int vi4 __attribute__((vector_size(16)));   // v4i32, builtin pointee type

union FragU { uint4 q[2]; v16bf v; };

// CDNA5 async global->LDS path (ASYNCcnt-tracked), if the toolchain has it.
#if defined(__has_builtin)
#if __has_builtin(__builtin_amdgcn_global_load_async_to_lds_b128)
#define HAVE_ASYNC_LDS 1
#endif
#endif
#ifndef HAVE_ASYNC_LDS
#define HAVE_ASYNC_LDS 0
#endif

__device__ __forceinline__ void copy16_to_lds(const unsigned short* g,
                                              unsigned short* l) {
#if HAVE_ASYNC_LDS
    __builtin_amdgcn_global_load_async_to_lds_b128(
        (__attribute__((address_space(1))) vi4*)(g),
        (__attribute__((address_space(3))) vi4*)(l), 0, 0);
#else
    *(uint4*)l = *(const uint4*)g;
#endif
}

__device__ __forceinline__ void wait_async_lds() {
#if HAVE_ASYNC_LDS
#if __has_builtin(__builtin_amdgcn_s_wait_asynccnt)
    __builtin_amdgcn_s_wait_asynccnt(0);
#else
    asm volatile("s_wait_asynccnt 0x0" ::: "memory");
#endif
#endif
}

__device__ __forceinline__ unsigned short f32_to_bf16(float f) {
    unsigned int u = __float_as_uint(f);
    unsigned int r = (u + 0x7FFFu + ((u >> 16) & 1u)) >> 16;   // RNE
    return (unsigned short)r;
}
__device__ __forceinline__ float bf16_to_f32(unsigned short h) {
    return __uint_as_float(((unsigned int)h) << 16);
}

// Load a 16x32 (A) or 32x16 (B) bf16 WMMA fragment.
// rowPtr points at the lane's row start + k-window base; per CDNA5 layout the
// lane's 16 elements are the two contiguous 8-half chunks at 8*lh and 16+8*lh.
__device__ __forceinline__ v16bf load_frag(const unsigned short* rowPtr, int lh) {
    FragU u;
    u.q[0] = *(const uint4*)(rowPtr + 8 * lh);
    u.q[1] = *(const uint4*)(rowPtr + 16 + 8 * lh);
    return u.v;
}

__device__ __forceinline__ v8f wmma_bf16(v16bf a, v16bf b, v8f c) {
    return __builtin_amdgcn_wmma_f32_16x16x32_bf16(false, a, false, b,
                                                   (short)0, c, false, false);
}

// ---------------------------------------------------------------------------
// f32 -> bf16 conversion (grid-stride)
// ---------------------------------------------------------------------------
__global__ __launch_bounds__(256)
void cvt_bf16_kernel(const float* __restrict__ in, unsigned short* __restrict__ out,
                     size_t n) {
    size_t i = (size_t)blockIdx.x * blockDim.x + threadIdx.x;
    size_t stride = (size_t)gridDim.x * blockDim.x;
    for (; i < n; i += stride) out[i] = f32_to_bf16(in[i]);
}

// ---------------------------------------------------------------------------
// Tiled WMMA GEMM:  C[M,N] = A[M,K] * B[N,K]^T   (both bf16, K contiguous)
// BM=BN=128, BK=64. 8 waves; wave (wm,wn) owns rows [wm*32,+32) x cols [wn*64,+64).
// LDS rows padded to 72 halves (144B): 16B-aligned, conflict-free frag loads.
// 16 WMMAs per barrier pair; tiles staged via async global->LDS.
// ---------------------------------------------------------------------------
template <int WRITE_BF16>
__global__ __launch_bounds__(256)
void gemm_bf16_nt_kernel(const unsigned short* __restrict__ A,
                         const unsigned short* __restrict__ Bm,
                         void* __restrict__ Cout,
                         int M, int N, int K, int ldc) {
    constexpr int BM = 128, BN = 128, BK = 64, LDSS = 72;
    __shared__ unsigned short As[BM * LDSS];
    __shared__ unsigned short Bs[BN * LDSS];

    const int bm = blockIdx.y * BM;
    const int bn = blockIdx.x * BN;
    const int tid = threadIdx.x;
    const int wave = tid >> 5, lane = tid & 31;
    const int lm = lane & 15, lh = lane >> 4;
    const int wm = wave >> 1, wn = wave & 1;

    v8f acc[2][4] = {};

    for (int k0 = 0; k0 < K; k0 += BK) {
        __syncthreads();
        // Cooperative tile staging: 128 rows x 128B = 1024 16B-chunks per tile.
        #pragma unroll
        for (int r = 0; r < 4; ++r) {
            int u = tid + r * 256;
            int row = u >> 3, ch = u & 7;
            copy16_to_lds(&A[(size_t)(bm + row) * K + k0 + ch * 8],
                          &As[row * LDSS + ch * 8]);
            copy16_to_lds(&Bm[(size_t)(bn + row) * K + k0 + ch * 8],
                          &Bs[row * LDSS + ch * 8]);
        }
        if (k0 + BK < K) {   // prefetch next K-slab into cache
            __builtin_prefetch(&A[(size_t)(bm + (tid >> 1)) * K + k0 + BK], 0, 0);
            __builtin_prefetch(&Bm[(size_t)(bn + (tid >> 1)) * K + k0 + BK], 0, 0);
        }
        wait_async_lds();
        __syncthreads();

        #pragma unroll
        for (int ks = 0; ks < 2; ++ks) {           // two K=32 sub-steps
            v16bf af[2], bfg[4];
            #pragma unroll
            for (int i = 0; i < 2; ++i)
                af[i] = load_frag(&As[(wm * 32 + i * 16 + lm) * LDSS + ks * 32], lh);
            #pragma unroll
            for (int j = 0; j < 4; ++j)
                bfg[j] = load_frag(&Bs[(wn * 64 + j * 16 + lm) * LDSS + ks * 32], lh);

            #pragma unroll
            for (int i = 0; i < 2; ++i)
                #pragma unroll
                for (int j = 0; j < 4; ++j)
                    acc[i][j] = wmma_bf16(af[i], bfg[j], acc[i][j]);
        }
    }

    // Epilogue. C/D layout: lane holds col n = lane&15; VGPR r holds row r+8*lh.
    #pragma unroll
    for (int i = 0; i < 2; ++i)
        #pragma unroll
        for (int j = 0; j < 4; ++j)
            #pragma unroll
            for (int r = 0; r < 8; ++r) {
                int m = bm + wm * 32 + i * 16 + r + 8 * lh;
                int n = bn + wn * 64 + j * 16 + lm;
                float v = acc[i][j][r];
                if (WRITE_BF16)
                    ((unsigned short*)Cout)[(size_t)m * ldc + n] = f32_to_bf16(v);
                else
                    ((float*)Cout)[(size_t)m * ldc + n] = v;
            }
}

// ---------------------------------------------------------------------------
// RoPE applied in place to Q and K slices of the QKV buffer (bf16).
// One thread per (b, t, head, j<32) rotation pair.
// ---------------------------------------------------------------------------
__global__ __launch_bounds__(256)
void rope_kernel(unsigned short* __restrict__ qkv,
                 const float* __restrict__ cosT,
                 const float* __restrict__ sinT) {
    size_t idx = (size_t)blockIdx.x * blockDim.x + threadIdx.x;
    const size_t total = (size_t)BB * TT * (HH + HKV) * 32;
    if (idx >= total) return;
    int j = idx & 31; size_t i = idx >> 5;
    int head = (int)(i % (HH + HKV)); i /= (HH + HKV);
    int t = (int)(i & (TT - 1));
    int b = (int)(i >> 11);

    size_t base = ((size_t)(b * TT + t)) * NQKV;
    base += (head < HH) ? (size_t)head * DH : (size_t)DD + (size_t)(head - HH) * DH;

    float c1 = cosT[t * DH + j],      s1 = sinT[t * DH + j];
    float c2 = cosT[t * DH + 32 + j], s2 = sinT[t * DH + 32 + j];
    float t1 = bf16_to_f32(qkv[base + j]);
    float t2 = bf16_to_f32(qkv[base + 32 + j]);
    qkv[base + j]      = f32_to_bf16(t1 * c1 - t2 * s1);
    qkv[base + 32 + j] = f32_to_bf16(t2 * c2 + t1 * s2);
}

// ---------------------------------------------------------------------------
// Causal flash attention with GQA (rep = 4).
// Block = 8 waves; each wave owns 16 query rows (128 q-rows / block).
// Per 32-key tile: QK^T = 4 wmma, online softmax (shfl_xor 16-lane row
// reductions), P -> per-wave LDS (C-layout -> A-layout), PV = 4 wmma.
// ---------------------------------------------------------------------------
__global__ __launch_bounds__(256)
void flash_attn_kernel(const unsigned short* __restrict__ qkv,
                       unsigned short* __restrict__ Y) {
    constexpr int KST = 72;  // K-tile LDS row stride (halves): 144B, conflict-free
    constexpr int VST = 40;  // V^T / P row stride (halves): 80B, conflict-free
    __shared__ unsigned short Kt[32 * KST];
    __shared__ unsigned short Vt[64 * VST];
    __shared__ unsigned short Ps[8 * 16 * VST];

    const int bh = blockIdx.y;
    const int b = bh >> 4, h = bh & 15, hkv = h >> 2;
    const int qBase = blockIdx.x * 128;
    const int tid = threadIdx.x, wave = tid >> 5, lane = tid & 31;
    const int lm = lane & 15, lh = lane >> 4;
    const int qrow = qBase + wave * 16;   // wave's first query row

    // Preload Q fragments (A-layout rows come straight from global, d contiguous)
    const unsigned short* qp =
        qkv + ((size_t)(b * TT + qrow + lm)) * NQKV + h * DH;
    v16bf q0 = load_frag(qp, lh);        // d = 0..31
    v16bf q1 = load_frag(qp + 32, lh);   // d = 32..63

    float m_i[8], l_i[8];
    v8f o[4] = {};
    #pragma unroll
    for (int r = 0; r < 8; ++r) { m_i[r] = -3.0e38f; l_i[r] = 0.0f; }

    const int sEnd = qBase + 128;
    for (int s0 = 0; s0 < sEnd; s0 += 32) {
        __syncthreads();
        {   // cooperative K / V^T staging: thread -> (key = tid>>3, ch = tid&7)
            int key = tid >> 3, ch = tid & 7;
            size_t rowb = ((size_t)(b * TT + s0 + key)) * NQKV;
            copy16_to_lds(&qkv[rowb + DD + hkv * DH + ch * 8],
                          &Kt[key * KST + ch * 8]);
            uint4 vv = *(const uint4*)&qkv[rowb + DD + HKV * DH + hkv * DH + ch * 8];
            unsigned short vt[8]; *(uint4*)vt = vv;
            #pragma unroll
            for (int j = 0; j < 8; ++j)
                Vt[(ch * 8 + j) * VST + key] = vt[j];   // transpose into [d][key]
        }
        wait_async_lds();
        __syncthreads();

        if (s0 <= qrow + 15) {  // wave-uniform: skip tiles fully above diagonal
            // ---- scores S = Q K^T  (two 16-key column tiles) ----
            v8f s[2] = {};
            #pragma unroll
            for (int nt = 0; nt < 2; ++nt) {
                const unsigned short* kr = &Kt[(nt * 16 + lm) * KST];
                s[nt] = wmma_bf16(q0, load_frag(kr, lh), s[nt]);
                s[nt] = wmma_bf16(q1, load_frag(kr + 32, lh), s[nt]);
            }
            // ---- scale + causal mask ----
            #pragma unroll
            for (int nt = 0; nt < 2; ++nt)
                #pragma unroll
                for (int r = 0; r < 8; ++r) {
                    float v = s[nt][r] * 0.125f;     // 1/sqrt(DH)
                    int scol = s0 + nt * 16 + lm;
                    int trow = qrow + r + 8 * lh;
                    s[nt][r] = (scol > trow) ? (v - 1.0e9f) : v;
                }
            // ---- online softmax (per-row over 16 lanes of each half) ----
            unsigned short* myP = &Ps[wave * 16 * VST];
            #pragma unroll
            for (int r = 0; r < 8; ++r) {
                float v = fmaxf(s[0][r], s[1][r]);
                v = fmaxf(v, __shfl_xor(v, 1, 32));
                v = fmaxf(v, __shfl_xor(v, 2, 32));
                v = fmaxf(v, __shfl_xor(v, 4, 32));
                v = fmaxf(v, __shfl_xor(v, 8, 32));
                float mn = fmaxf(m_i[r], v);
                float alpha = __expf(m_i[r] - mn);
                m_i[r] = mn;
                float p0 = __expf(s[0][r] - mn);
                float p1 = __expf(s[1][r] - mn);
                float sum = p0 + p1;
                sum += __shfl_xor(sum, 1, 32);
                sum += __shfl_xor(sum, 2, 32);
                sum += __shfl_xor(sum, 4, 32);
                sum += __shfl_xor(sum, 8, 32);
                l_i[r] = l_i[r] * alpha + sum;
                #pragma unroll
                for (int nt2 = 0; nt2 < 4; ++nt2) o[nt2][r] *= alpha;
                // stash P (C-layout -> LDS row-major 16x32)
                int prow = r + 8 * lh;
                myP[prow * VST + lm]      = f32_to_bf16(p0);
                myP[prow * VST + 16 + lm] = f32_to_bf16(p1);
            }
            // same-wave LDS store -> load ordering
            asm volatile("s_wait_dscnt 0x0" ::: "memory");
            v16bf pf = load_frag(&myP[lm * VST], lh);   // A-layout 16x32
            #pragma unroll
            for (int nt2 = 0; nt2 < 4; ++nt2) {
                v16bf bv = load_frag(&Vt[(nt2 * 16 + lm) * VST], lh);
                o[nt2] = wmma_bf16(pf, bv, o[nt2]);
            }
        }
    }

    // ---- normalize and store Y (bf16) ----
    #pragma unroll
    for (int nt2 = 0; nt2 < 4; ++nt2)
        #pragma unroll
        for (int r = 0; r < 8; ++r) {
            int t = qrow + r + 8 * lh;
            int d = h * DH + nt2 * 16 + lm;
            float v = o[nt2][r] / l_i[r];
            Y[((size_t)(b * TT + t)) * DD + d] = f32_to_bf16(v);
        }
}

// ---------------------------------------------------------------------------
// Host-side orchestration
// ---------------------------------------------------------------------------
extern "C" void kernel_launch(void* const* d_in, const int* in_sizes, int n_in,
                              void* d_out, int out_size, void* d_ws, size_t ws_size,
                              hipStream_t stream) {
    const float* x        = (const float*)d_in[0];
    const float* rope_cos = (const float*)d_in[1];
    const float* rope_sin = (const float*)d_in[2];
    // d_in[3] = attn_mask (pure causal -1e9 triu; reproduced analytically)
    const float* Wq = (const float*)d_in[4];
    const float* Wk = (const float*)d_in[5];
    const float* Wv = (const float*)d_in[6];
    const float* Wo = (const float*)d_in[7];

    // Workspace layout (bf16 buffers), 256B-aligned partitions
    uint8_t* ws = (uint8_t*)d_ws;
    size_t off = 0;
    auto carve = [&](size_t bytes) {
        uint8_t* p = ws + off;
        off += (bytes + 255) & ~(size_t)255;
        return p;
    };
    unsigned short* xb   = (unsigned short*)carve((size_t)MROWS * DD * 2);
    unsigned short* wqkv = (unsigned short*)carve((size_t)NQKV * DD * 2);
    unsigned short* wo   = (unsigned short*)carve((size_t)DD * DD * 2);
    unsigned short* qkv  = (unsigned short*)carve((size_t)MROWS * NQKV * 2);
    unsigned short* yb   = (unsigned short*)carve((size_t)MROWS * DD * 2);

    // 1) precision conversion
    cvt_bf16_kernel<<<2048, 256, 0, stream>>>(x, xb, (size_t)MROWS * DD);
    cvt_bf16_kernel<<<1024, 256, 0, stream>>>(Wq, wqkv, (size_t)HH * DH * DD);
    cvt_bf16_kernel<<<256, 256, 0, stream>>>(Wk, wqkv + (size_t)DD * DD,
                                             (size_t)HKV * DH * DD);
    cvt_bf16_kernel<<<256, 256, 0, stream>>>(Wv, wqkv + (size_t)(DD + HKV * DH) * DD,
                                             (size_t)HKV * DH * DD);
    cvt_bf16_kernel<<<1024, 256, 0, stream>>>(Wo, wo, (size_t)DD * DD);

    // 2) fused QKV projection: [4096,1536] = xb[4096,1024] * Wqkv[1536,1024]^T
    {
        dim3 grid(NQKV / 128, MROWS / 128);
        gemm_bf16_nt_kernel<1><<<grid, 256, 0, stream>>>(xb, wqkv, qkv,
                                                         MROWS, NQKV, DD, NQKV);
    }
    // 3) RoPE on Q and K slices
    {
        size_t total = (size_t)BB * TT * (HH + HKV) * 32;
        rope_kernel<<<(unsigned)((total + 255) / 256), 256, 0, stream>>>(
            qkv, rope_cos, rope_sin);
    }
    // 4) causal flash attention -> yb[4096,1024]
    {
        dim3 grid(TT / 128, BB * HH);
        flash_attn_kernel<<<grid, 256, 0, stream>>>(qkv, yb);
    }
    // 5) output projection: out[4096,1024] = yb * Wo^T (f32 out)
    {
        dim3 grid(DD / 128, MROWS / 128);
        gemm_bf16_nt_kernel<0><<<grid, 256, 0, stream>>>(yb, wo, (float*)d_out,
                                                         MROWS, DD, DD, DD);
    }
}
